// GATEncoder_75814762709160
// MI455X (gfx1250) — matrix operand order repaired and checked by reading, not compiled, not run on hardware
//
#include <hip/hip_runtime.h>

typedef _Float16 half_t;
typedef __attribute__((ext_vector_type(16))) _Float16 v16h;
typedef __attribute__((ext_vector_type(8)))  float    v8f;

#define NEG_SLOPE 0.2f

// ---------- helpers ----------

// WMMA 16-bit A-matrix 16x32 layout (ISA 7.12.2): half i of lane l maps to
//   row M = l%16,  K = (i%8) + 16*(i/8) + 8*(l/16)
// Inverse (used when staging): column k (0..31), row r ->
//   lane = 16*((k>>3)&1) + r,  i = (k&7) | (((k>>4)&1)<<3)
// Consecutive even/odd k land in consecutive halves of one lane, so we stage
// pairs as packed dwords: u-index = ((k&7)>>1) | (((k>>4)&1)<<2).

__device__ __forceinline__ unsigned pack2h(float x, float y) {
    half_t lo = (half_t)x, hi = (half_t)y;
    return ((unsigned)__builtin_bit_cast(unsigned short, hi) << 16) |
           (unsigned)__builtin_bit_cast(unsigned short, lo);
}

// monotone float->uint key so atomicMax(u32) == float max
__device__ __forceinline__ unsigned f2key(float f) {
    unsigned u = __float_as_uint(f);
    return (u & 0x80000000u) ? ~u : (u | 0x80000000u);
}
__device__ __forceinline__ float key2f(unsigned k) {
    unsigned u = (k & 0x80000000u) ? (k & 0x7FFFFFFFu) : ~k;
    return __uint_as_float(u);
}

// ---------- init kernels ----------

__global__ void zero_kernel(unsigned* __restrict__ p, long count) {
    long i = (long)blockIdx.x * blockDim.x + threadIdx.x;
    if (i < count) p[i] = 0u;
}

// Convert f32 weights to f16 *pre-permuted into WMMA B-fragment order*:
//   frag[(colt*KK + kkIdx)*32*16 + lane*16 + i] =
//       W[(kkIdx*32 + 8*(lane/16) + (i%8) + 16*(i/8)) * NCOLS + colt*16 + (lane%16)]
// so the GEMM reads each B fragment as one contiguous v16h (2x b128).
__global__ void pack_weights_kernel(const float* __restrict__ w1l,
                                    const float* __restrict__ w1r,
                                    const float* __restrict__ w2l,
                                    const float* __restrict__ w2r,
                                    half_t* __restrict__ f1l,
                                    half_t* __restrict__ f1r,
                                    half_t* __restrict__ f2l,
                                    half_t* __restrict__ f2r) {
    int idx = blockIdx.x * blockDim.x + threadIdx.x;
    if (idx < 4 * 4 * 32 * 16) {   // W1: 128x64, 4 coltiles x 4 K-steps
        int i = idx & 15, lane = (idx >> 4) & 31;
        int kkIdx = (idx >> 9) & 3, colt = idx >> 11;
        int k = kkIdx * 32 + ((lane >> 4) << 3) + (i & 7) + ((i >> 3) << 4);
        int col = colt * 16 + (lane & 15);
        f1l[idx] = (half_t)w1l[k * 64 + col];
        f1r[idx] = (half_t)w1r[k * 64 + col];
    }
    if (idx < 2 * 32 * 16) {       // W2: 64x16, 1 coltile x 2 K-steps
        int i = idx & 15, lane = (idx >> 4) & 31;
        int kkIdx = (idx >> 9) & 1;
        int k = kkIdx * 32 + ((lane >> 4) << 3) + (i & 7) + ((i >> 3) << 4);
        int col = lane & 15;
        f2l[idx] = (half_t)w2l[k * 16 + col];
        f2r[idx] = (half_t)w2r[k * 16 + col];
    }
}

// ---------- WMMA GEMMs ----------

// Y = X[Nn,128] @ W[128,64] for two weight matrices (xl, xr).
// Block: 256 threads = 8 waves, one 16-row tile; wave w -> 16 output columns
// (waves 0..3 -> xl, 4..7 -> xr). X tile staged in LDS in A-fragment order so
// each fragment is a contiguous 32B LDS read; B fragments are contiguous 32B
// global reads of the pre-permuted f16 weights (L2 resident).
__global__ __launch_bounds__(256) void gemm1_kernel(
    const float* __restrict__ X,
    const half_t* __restrict__ WfL, const half_t* __restrict__ WfR,
    float* __restrict__ xl, float* __restrict__ xr, int Nn)
{
    __shared__ __align__(32) unsigned sxA[4][32][8];   // [kkIdx][lane][pair]
    const int tid = threadIdx.x;
    const int rowbase = blockIdx.x * 16;
    const bool full = (rowbase + 16 <= Nn);
#pragma unroll
    for (int j = 0; j < 4; ++j) {
        int p = tid + 256 * j;              // pair index 0..1023
        int r = p >> 6, c = (p & 63) * 2;   // row 0..15, col 0..126 even
        int row = rowbase + r;
        float2 v = make_float2(0.f, 0.f);
        if (full || row < Nn) v = *(const float2*)(X + (size_t)row * 128 + c);
        int kkIdx = c >> 5;
        int lane  = (((c >> 3) & 1) << 4) | r;
        int ui    = ((c & 7) >> 1) | (((c >> 4) & 1) << 2);
        sxA[kkIdx][lane][ui] = pack2h(v.x, v.y);
    }
    __syncthreads();

    const int wave = tid >> 5, lane = tid & 31;
    const half_t* Wf = (wave < 4) ? WfL : WfR;
    float* out = (wave < 4) ? xl : xr;
    const int colt = wave & 3;
    v8f acc = {};
#pragma unroll
    for (int kkIdx = 0; kkIdx < 4; ++kkIdx) {
        v16h a = *(const v16h*)&sxA[kkIdx][lane][0];
        v16h b = *(const v16h*)(Wf + (size_t)((colt * 4 + kkIdx) * 32 + lane) * 16);
        acc = __builtin_amdgcn_wmma_f32_16x16x32_f16(false, a, false, b,
                                                     (short)0, acc, false, false);
    }
    const int n = lane & 15;
    const int mhi = ((lane >> 4) & 1) << 3;
    if (full) {
#pragma unroll
        for (int r = 0; r < 8; ++r)
            out[(size_t)(rowbase + r + mhi) * 64 + colt * 16 + n] = acc[r];
    } else {
#pragma unroll
        for (int r = 0; r < 8; ++r) {
            int row = rowbase + r + mhi;
            if (row < Nn) out[(size_t)row * 64 + colt * 16 + n] = acc[r];
        }
    }
}

// Z = H[Nn,64] @ W[64,16] for two weight matrices (zl, zr).
// Block: 64 threads = 2 waves (wave0 -> zl, wave1 -> zr), K-loop of 2 WMMAs.
__global__ __launch_bounds__(64) void gemm2_kernel(
    const float* __restrict__ Hm,
    const half_t* __restrict__ WfL, const half_t* __restrict__ WfR,
    float* __restrict__ zl, float* __restrict__ zr, int Nn)
{
    __shared__ __align__(32) unsigned sxA[2][32][8];
    const int tid = threadIdx.x;
    const int rowbase = blockIdx.x * 16;
    const bool full = (rowbase + 16 <= Nn);
#pragma unroll
    for (int j = 0; j < 8; ++j) {
        int p = tid + 64 * j;               // pair index 0..511
        int r = p >> 5, c = (p & 31) * 2;   // row 0..15, col 0..62 even
        int row = rowbase + r;
        float2 v = make_float2(0.f, 0.f);
        if (full || row < Nn) v = *(const float2*)(Hm + (size_t)row * 64 + c);
        int kkIdx = c >> 5;
        int lane  = (((c >> 3) & 1) << 4) | r;
        int ui    = ((c & 7) >> 1) | (((c >> 4) & 1) << 2);
        sxA[kkIdx][lane][ui] = pack2h(v.x, v.y);
    }
    __syncthreads();

    const int wave = tid >> 5, lane = tid & 31;
    const half_t* Wf = wave ? WfR : WfL;
    float* out = wave ? zr : zl;
    v8f acc = {};
#pragma unroll
    for (int kkIdx = 0; kkIdx < 2; ++kkIdx) {
        v16h a = *(const v16h*)&sxA[kkIdx][lane][0];
        v16h b = *(const v16h*)(Wf + (size_t)(kkIdx * 32 + lane) * 16);
        acc = __builtin_amdgcn_wmma_f32_16x16x32_f16(false, a, false, b,
                                                     (short)0, acc, false, false);
    }
    const int n = lane & 15;
    const int mhi = ((lane >> 4) & 1) << 3;
    if (full) {
#pragma unroll
        for (int r = 0; r < 8; ++r)
            out[(size_t)(rowbase + r + mhi) * 16 + n] = acc[r];
    } else {
#pragma unroll
        for (int r = 0; r < 8; ++r) {
            int row = rowbase + r + mhi;
            if (row < Nn) out[(size_t)row * 16 + n] = acc[r];
        }
    }
}

// ---------- edge-wise kernels (memory/atomic bound) ----------
// Edge id gid in [0, E+Nn): gid < E -> (src=ei[gid], dst=ei[E+gid]);
// gid >= E -> self loop (src=dst=gid-E).
// L lanes cooperate per edge; each lane handles 4 features (float4).

template <int H, int L>
__global__ __launch_bounds__(256) void edge_score_kernel(
    const float* __restrict__ xl, const float* __restrict__ xr,
    const float* __restrict__ att, const int* __restrict__ ei,
    int E, int Nn, float* __restrict__ ew, unsigned* __restrict__ mkey)
{
    const long t = (long)blockIdx.x * 256 + threadIdx.x;
    const long gid = t / L;
    const int li = (int)(t % L);
    const long Etot = (long)E + Nn;
    if (gid >= Etot) return;
    int s, d;
    if (gid < E) { s = ei[gid]; d = ei[(long)E + gid]; }
    else         { s = d = (int)(gid - E); }

    const float4 xv = *(const float4*)(xl + (size_t)s * H + li * 4);
    const float4 rv = *(const float4*)(xr + (size_t)d * H + li * 4);
    const float4 av = *(const float4*)(att + li * 4);
    float p = 0.f, tv;
    tv = xv.x + rv.x; tv = tv > 0.f ? tv : tv * NEG_SLOPE; p += av.x * tv;
    tv = xv.y + rv.y; tv = tv > 0.f ? tv : tv * NEG_SLOPE; p += av.y * tv;
    tv = xv.z + rv.z; tv = tv > 0.f ? tv : tv * NEG_SLOPE; p += av.z * tv;
    tv = xv.w + rv.w; tv = tv > 0.f ? tv : tv * NEG_SLOPE; p += av.w * tv;
#pragma unroll
    for (int off = L >> 1; off > 0; off >>= 1) p += __shfl_xor(p, off, L);
    if (li == 0) {
        ew[gid] = p;
        atomicMax(mkey + d, f2key(p));
    }
}

__global__ __launch_bounds__(256) void edge_softmax_kernel(
    const int* __restrict__ ei, int E, int Nn,
    const unsigned* __restrict__ mkey, float* __restrict__ ew,
    float* __restrict__ denom)
{
    const long gid = (long)blockIdx.x * 256 + threadIdx.x;
    const long Etot = (long)E + Nn;
    if (gid >= Etot) return;
    const int d = (gid < E) ? ei[(long)E + gid] : (int)(gid - E);
    const float w = __expf(ew[gid] - key2f(mkey[d]));
    ew[gid] = w;
    unsafeAtomicAdd(denom + d, w);
}

template <int H, int L>
__global__ __launch_bounds__(256) void edge_aggregate_kernel(
    const float* __restrict__ xl, const int* __restrict__ ei,
    int E, int Nn, const float* __restrict__ ew,
    const float* __restrict__ denom, float* __restrict__ acc)
{
    const long t = (long)blockIdx.x * 256 + threadIdx.x;
    const long gid = t / L;
    const int li = (int)(t % L);
    const long Etot = (long)E + Nn;
    if (gid >= Etot) return;
    int s, d;
    if (gid < E) { s = ei[gid]; d = ei[(long)E + gid]; }
    else         { s = d = (int)(gid - E); }
    const float alpha = ew[gid] / denom[d];
    const float4 xv = *(const float4*)(xl + (size_t)s * H + li * 4);
    float* o = acc + (size_t)d * H + li * 4;
    unsafeAtomicAdd(o + 0, alpha * xv.x);
    unsafeAtomicAdd(o + 1, alpha * xv.y);
    unsafeAtomicAdd(o + 2, alpha * xv.z);
    unsafeAtomicAdd(o + 3, alpha * xv.w);
}

// ---------- pointwise epilogues ----------

__global__ __launch_bounds__(256) void bias_elu_kernel(
    const float* __restrict__ acc, const float* __restrict__ b,
    float* __restrict__ h, long total)
{
    const long i = (long)blockIdx.x * 256 + threadIdx.x;
    if (i >= total) return;
    const float v = acc[i] + b[i & 63];
    h[i] = v > 0.f ? v : (__expf(v) - 1.f);
}

__global__ __launch_bounds__(256) void bias_softmax_kernel(
    const float* __restrict__ acc, const float* __restrict__ b,
    float* __restrict__ out, int Nn)
{
    const int nidx = blockIdx.x * 256 + threadIdx.x;
    if (nidx >= Nn) return;
    float z[16];
    float mx = -3.0e38f;
#pragma unroll
    for (int k = 0; k < 16; ++k) {
        z[k] = acc[(size_t)nidx * 16 + k] + b[k];
        mx = fmaxf(mx, z[k]);
    }
    float ssum = 0.f;
#pragma unroll
    for (int k = 0; k < 16; ++k) { z[k] = __expf(z[k] - mx); ssum += z[k]; }
    const float inv = 1.f / ssum;
#pragma unroll
    for (int k = 0; k < 16; ++k) out[(size_t)nidx * 16 + k] = z[k] * inv;
}

// ---------- host-side orchestration ----------

static inline size_t alloc_off(size_t& off, size_t bytes) {
    size_t o = off;
    off = (off + bytes + 255) & ~(size_t)255;
    return o;
}

extern "C" void kernel_launch(void* const* d_in, const int* in_sizes, int n_in,
                              void* d_out, int out_size, void* d_ws, size_t ws_size,
                              hipStream_t stream) {
    const float* X   = (const float*)d_in[0];
    const int*   ei  = (const int*)  d_in[1];
    // d_in[2] = batch (unused)
    const float* W1l = (const float*)d_in[3];
    const float* W1r = (const float*)d_in[4];
    const float* a1  = (const float*)d_in[5];
    const float* b1  = (const float*)d_in[6];
    const float* W2l = (const float*)d_in[7];
    const float* W2r = (const float*)d_in[8];
    const float* a2  = (const float*)d_in[9];
    const float* b2  = (const float*)d_in[10];
    float* out = (float*)d_out;

    const int Nn = in_sizes[0] / 128;
    const int E  = in_sizes[1] / 2;
    const long Etot = (long)E + Nn;

    char* base = (char*)d_ws;
    size_t off = 0;
    half_t* f1l = (half_t*)(base + alloc_off(off, 128 * 64 * sizeof(half_t)));
    half_t* f1r = (half_t*)(base + alloc_off(off, 128 * 64 * sizeof(half_t)));
    half_t* f2l = (half_t*)(base + alloc_off(off, 64 * 16 * sizeof(half_t)));
    half_t* f2r = (half_t*)(base + alloc_off(off, 64 * 16 * sizeof(half_t)));
    float* xl1  = (float*)(base + alloc_off(off, (size_t)Nn * 64 * 4));
    float* xr1  = (float*)(base + alloc_off(off, (size_t)Nn * 64 * 4));
    float* hbuf = (float*)(base + alloc_off(off, (size_t)Nn * 64 * 4));
    float* zl2  = (float*)(base + alloc_off(off, (size_t)Nn * 16 * 4));
    float* zr2  = (float*)(base + alloc_off(off, (size_t)Nn * 16 * 4));
    float* ew   = (float*)(base + alloc_off(off, (size_t)Etot * 4));
    // zero-initialized region (accumulators + softmax state, both layers)
    const size_t zstart = off;
    float*    acc1 = (float*)   (base + alloc_off(off, (size_t)Nn * 64 * 4));
    float*    acc2 = (float*)   (base + alloc_off(off, (size_t)Nn * 16 * 4));
    unsigned* mk1  = (unsigned*)(base + alloc_off(off, (size_t)Nn * 4));
    float*    dn1  = (float*)   (base + alloc_off(off, (size_t)Nn * 4));
    unsigned* mk2  = (unsigned*)(base + alloc_off(off, (size_t)Nn * 4));
    float*    dn2  = (float*)   (base + alloc_off(off, (size_t)Nn * 4));
    (void)ws_size; (void)n_in; (void)out_size;

    const long zcount = (long)((off - zstart) / 4);
    zero_kernel<<<(int)((zcount + 255) / 256), 256, 0, stream>>>(
        (unsigned*)(base + zstart), zcount);

    pack_weights_kernel<<<(128 * 64 + 255) / 256, 256, 0, stream>>>(
        W1l, W1r, W2l, W2r, f1l, f1r, f2l, f2r);

    // ---- layer 1 ----
    const int mtiles = (Nn + 15) / 16;
    gemm1_kernel<<<mtiles, 256, 0, stream>>>(X, f1l, f1r, xl1, xr1, Nn);

    {
        const long t1 = Etot * 16;
        edge_score_kernel<64, 16><<<(int)((t1 + 255) / 256), 256, 0, stream>>>(
            xl1, xr1, a1, ei, E, Nn, ew, mk1);
        edge_softmax_kernel<<<(int)((Etot + 255) / 256), 256, 0, stream>>>(
            ei, E, Nn, mk1, ew, dn1);
        edge_aggregate_kernel<64, 16><<<(int)((t1 + 255) / 256), 256, 0, stream>>>(
            xl1, ei, E, Nn, ew, dn1, acc1);
    }
    {
        const long total = (long)Nn * 64;
        bias_elu_kernel<<<(int)((total + 255) / 256), 256, 0, stream>>>(
            acc1, b1, hbuf, total);
    }

    // ---- layer 2 ----
    gemm2_kernel<<<mtiles, 64, 0, stream>>>(hbuf, f2l, f2r, zl2, zr2, Nn);

    {
        const long t2 = Etot * 4;
        edge_score_kernel<16, 4><<<(int)((t2 + 255) / 256), 256, 0, stream>>>(
            zl2, zr2, a2, ei, E, Nn, ew, mk2);
        edge_softmax_kernel<<<(int)((Etot + 255) / 256), 256, 0, stream>>>(
            ei, E, Nn, mk2, ew, dn2);
        edge_aggregate_kernel<16, 4><<<(int)((t2 + 255) / 256), 256, 0, stream>>>(
            zl2, ei, E, Nn, ew, dn2, acc2);
    }
    bias_softmax_kernel<<<(Nn + 255) / 256, 256, 0, stream>>>(acc2, b2, out, Nn);
}
